// CapsNet_FOOD101_12601434046783
// MI455X (gfx1250) — compile-verified, matrix operand order checked by reading, not run on hardware
//
#include <hip/hip_runtime.h>
#include <hip/hip_bf16.h>

typedef __attribute__((ext_vector_type(16))) _Float16 v16h;
typedef __attribute__((ext_vector_type(8)))  float    v8f;

union HFrag {
  v16h  v;
  _Float16 h[16];
  uint4 q[2];
};

static __device__ __forceinline__ v8f wmma_f16(const HFrag& a, const HFrag& b, v8f c) {
  // D = A(16x32 f16) * B(32x16 f16) + C(16x16 f32)
  return __builtin_amdgcn_wmma_f32_16x16x32_f16(
      /*neg_a=*/false, a.v, /*neg_b=*/false, b.v,
      /*c_mod=*/(short)0, c, /*reuse_a=*/false, /*reuse_b=*/false);
}

// ---------------- problem constants ----------------
constexpr int BATCH = 32;
constexpr int OC1 = 256, K1 = 363, K1P = 384, OH1 = 34;   // conv1: 3x112x112 -> 256x34x34, k=11,s=3
constexpr int OC2 = 256, K2 = 20736, OH2 = 9;             // pconv: 256x34x34 -> 256x9x9,  k=9, s=3
constexpr int NCAPS = 2592, DIM = 8, OCAPS = 101, VDIM = 16;
constexpr int N1 = BATCH * OH1 * OH1;     // 36992 = 2312*16
constexpr int N2 = BATCH * OH2 * OH2;     // 2592  = 162*16
constexpr int NT1 = N1 / 16, NT2 = N2 / 16;
constexpr int ITILES = NCAPS / 4;         // 648 K-tiles of 4 capsules in v-pass
constexpr int VCHUNKS = 24;               // 648 = 24 * 27
constexpr int VT_PER_CHUNK = ITILES / VCHUNKS;
constexpr int BCHUNKS = 16;               // 2592 = 16 * 162
constexpr int I_PER_BCHUNK = NCAPS / BCHUNKS;
constexpr int BO = BATCH * OCAPS;         // 3232
constexpr int BI = BATCH * NCAPS;         // 82944

// ---------------- small utility kernels ----------------
__global__ void k_f32_to_f16(const float* __restrict__ src, _Float16* __restrict__ dst, int n) {
  for (int i = blockIdx.x * blockDim.x + threadIdx.x; i < n; i += gridDim.x * blockDim.x)
    dst[i] = (_Float16)src[i];
}

__global__ void k_cvt_w1(const float* __restrict__ w, _Float16* __restrict__ dst) {
  int tid = blockIdx.x * blockDim.x + threadIdx.x;
  if (tid >= OC1 * K1P) return;
  int oc = tid / K1P, k = tid % K1P;
  dst[tid] = (k < K1) ? (_Float16)w[oc * K1 + k] : (_Float16)0.0f;
}

__global__ void k_zero_f32(float* __restrict__ p, int n) {
  for (int i = blockIdx.x * blockDim.x + threadIdx.x; i < n; i += gridDim.x * blockDim.x)
    p[i] = 0.0f;
}

// im2col K -> memory-offset tables (computed once; kills all div/mods in GEMM loops)
__global__ void k_make_offs(int* __restrict__ offs1, int* __restrict__ offs2) {
  const int tid = blockIdx.x * blockDim.x + threadIdx.x;
  if (tid < K1P) {
    if (tid < K1) {
      const int c = tid / 121, r = tid % 121, kh = r / 11, kw = r % 11;
      offs1[tid] = c * (112 * 112) + kh * 112 + kw;
    } else {
      offs1[tid] = 0;   // A (weights) is zero-padded here; B value is don't-care (finite)
    }
  } else if (tid < K1P + K2) {
    const int k = tid - K1P;
    const int c = k / 81, r = k % 81, kh = r / 9, kw = r % 9;
    offs2[k] = c * (OH1 * OH1) + kh * OH1 + kw;
  }
}

// ---------------- conv1 implicit GEMM (WMMA), output stored f16 ----------------
// Each wave computes 4 M-tiles (64 out-channels) for one shared N-tile: the 16-element
// im2col B gather is amortized over 4 WMMAs. grid = NT1 blocks, block = 128 (4 waves).
__global__ void k_conv1(const float* __restrict__ x, const _Float16* __restrict__ w1h,
                        const float* __restrict__ bias, const int* __restrict__ offs,
                        _Float16* __restrict__ h16) {
  const int lane = threadIdx.x & 31;
  const int wav  = threadIdx.x >> 5;
  const int ntile = blockIdx.x;
  const int lm = lane & 15;
  const int hi = lane >> 4;
  // decode this lane's B column
  const int n  = ntile * 16 + lm;
  const int nb = n / (OH1 * OH1);
  const int nr = n % (OH1 * OH1);
  const int oh = nr / OH1, ow = nr % OH1;
  const float* xb = x + (size_t)nb * 3 * 112 * 112 + (oh * 3) * 112 + (ow * 3);
  const _Float16* ap[4];
#pragma unroll
  for (int mi = 0; mi < 4; ++mi)
    ap[mi] = w1h + (size_t)((wav * 4 + mi) * 16 + lm) * K1P;
  v8f acc0 = {}, acc1 = {}, acc2 = {}, acc3 = {};
  for (int k0 = 0; k0 < K1P; k0 += 32) {
    const int kb = k0 + hi * 8;
    HFrag B;
    const int* o0 = offs + kb;        // 8 consecutive ints -> vector loads
    const int* o1 = offs + kb + 16;
#pragma unroll
    for (int j = 0; j < 8; ++j) {
      B.h[j]     = (_Float16)xb[o0[j]];
      B.h[j + 8] = (_Float16)xb[o1[j]];
    }
    HFrag A0, A1, A2, A3;
    A0.q[0] = *(const uint4*)(ap[0] + kb); A0.q[1] = *(const uint4*)(ap[0] + kb + 16);
    A1.q[0] = *(const uint4*)(ap[1] + kb); A1.q[1] = *(const uint4*)(ap[1] + kb + 16);
    A2.q[0] = *(const uint4*)(ap[2] + kb); A2.q[1] = *(const uint4*)(ap[2] + kb + 16);
    A3.q[0] = *(const uint4*)(ap[3] + kb); A3.q[1] = *(const uint4*)(ap[3] + kb + 16);
    acc0 = wmma_f16(A0, B, acc0);
    acc1 = wmma_f16(A1, B, acc1);
    acc2 = wmma_f16(A2, B, acc2);
    acc3 = wmma_f16(A3, B, acc3);
  }
  v8f accs[4] = {acc0, acc1, acc2, acc3};
#pragma unroll
  for (int mi = 0; mi < 4; ++mi) {
#pragma unroll
    for (int r = 0; r < 8; ++r) {
      const int oc = (wav * 4 + mi) * 16 + r + 8 * hi;
      const float o = accs[mi][r] + bias[oc];
      h16[(((size_t)nb * OC1 + oc) * OH1 + oh) * OH1 + ow] = (_Float16)fmaxf(o, 0.0f);
    }
  }
}

// ---------------- primary-caps conv implicit GEMM (WMMA), f16 in / f32 out ----------------
// Same 4-Mtile-per-wave structure; B gather loads f16 activations directly (no converts).
__global__ void k_pconv(const _Float16* __restrict__ h16, const _Float16* __restrict__ w2h,
                        const float* __restrict__ bias, const int* __restrict__ offs,
                        float* __restrict__ p) {
  const int lane = threadIdx.x & 31;
  const int wav  = threadIdx.x >> 5;
  const int ntile = blockIdx.x;
  const int lm = lane & 15;
  const int hi = lane >> 4;
  const int n  = ntile * 16 + lm;
  const int nb = n / (OH2 * OH2);
  const int nr = n % (OH2 * OH2);
  const int oh = nr / OH2, ow = nr % OH2;
  const _Float16* hb = h16 + (size_t)nb * OC1 * OH1 * OH1 + (oh * 3) * OH1 + (ow * 3);
  const _Float16* ap[4];
#pragma unroll
  for (int mi = 0; mi < 4; ++mi)
    ap[mi] = w2h + (size_t)((wav * 4 + mi) * 16 + lm) * K2;
  v8f acc0 = {}, acc1 = {}, acc2 = {}, acc3 = {};
  for (int k0 = 0; k0 < K2; k0 += 32) {
    const int kb = k0 + hi * 8;
    HFrag B;
    const int* o0 = offs + kb;
    const int* o1 = offs + kb + 16;
#pragma unroll
    for (int j = 0; j < 8; ++j) {
      B.h[j]     = hb[o0[j]];
      B.h[j + 8] = hb[o1[j]];
    }
    HFrag A0, A1, A2, A3;
    A0.q[0] = *(const uint4*)(ap[0] + kb); A0.q[1] = *(const uint4*)(ap[0] + kb + 16);
    A1.q[0] = *(const uint4*)(ap[1] + kb); A1.q[1] = *(const uint4*)(ap[1] + kb + 16);
    A2.q[0] = *(const uint4*)(ap[2] + kb); A2.q[1] = *(const uint4*)(ap[2] + kb + 16);
    A3.q[0] = *(const uint4*)(ap[3] + kb); A3.q[1] = *(const uint4*)(ap[3] + kb + 16);
    acc0 = wmma_f16(A0, B, acc0);
    acc1 = wmma_f16(A1, B, acc1);
    acc2 = wmma_f16(A2, B, acc2);
    acc3 = wmma_f16(A3, B, acc3);
  }
  v8f accs[4] = {acc0, acc1, acc2, acc3};
#pragma unroll
  for (int mi = 0; mi < 4; ++mi) {
#pragma unroll
    for (int r = 0; r < 8; ++r) {
      const int oc = (wav * 4 + mi) * 16 + r + 8 * hi;
      p[(size_t)nb * K2 + (size_t)oc * 81 + oh * OH2 + ow] = accs[mi][r] + bias[oc];
    }
  }
}

// ---------------- squash primary capsules: p[B,20736] -> u[B,2592,8] (f32 + f16) ----------------
__global__ void k_squash_u(const float* __restrict__ p, float* __restrict__ u,
                           _Float16* __restrict__ uh) {
  const int tid = blockIdx.x * blockDim.x + threadIdx.x;
  if (tid >= BI) return;
  const float* pp = p + (size_t)tid * DIM;
  float t[DIM], sq = 0.0f;
#pragma unroll
  for (int d = 0; d < DIM; ++d) { t[d] = pp[d]; sq += t[d] * t[d]; }
  const float scale = sq / ((1.0f + sq) * sqrtf(sq + 1e-8f));
#pragma unroll
  for (int d = 0; d < DIM; ++d) {
    const float v = t[d] * scale;
    u[(size_t)tid * DIM + d]  = v;
    uh[(size_t)tid * DIM + d] = (_Float16)v;
  }
}

// ---------------- softmax of b_logits over o (axis 1) ----------------
__global__ void k_softmax_o(const float* __restrict__ blog, float* __restrict__ cmat) {
  const int tid = blockIdx.x * blockDim.x + threadIdx.x;
  if (tid >= BI) return;
  const int b = tid / NCAPS, i = tid % NCAPS;
  const float* base = blog + (size_t)b * OCAPS * NCAPS + i;
  float m = -1e30f;
  for (int o = 0; o < OCAPS; ++o) m = fmaxf(m, base[(size_t)o * NCAPS]);
  float s = 0.0f;
  for (int o = 0; o < OCAPS; ++o) s += expf(base[(size_t)o * NCAPS] - m);
  const float inv = 1.0f / s;
  float* cb = cmat + (size_t)b * OCAPS * NCAPS + i;
  for (int o = 0; o < OCAPS; ++o) cb[(size_t)o * NCAPS] = expf(base[(size_t)o * NCAPS] - m) * inv;
}

// ---------------- v-pass: s[b,o,v] = sum_i c[b,o,i] * x_hat[b,o,i,v], x_hat recomputed (WMMA)
// K-tile packs 4 capsules * 8 dims = 32; c folded into B so the WMMA accumulates the weighted sum.
// grid(OCAPS, VCHUNKS), block 64 (2 waves = 2 batch halves); deterministic partial buffers.
__global__ void k_vpass(const _Float16* __restrict__ Wh, const float* __restrict__ u,
                        const float* __restrict__ cmat, float* __restrict__ spart) {
  const int lane  = threadIdx.x & 31;
  const int bhalf = threadIdx.x >> 5;
  const int o = blockIdx.x, chunk = blockIdx.y;
  const int lm = lane & 15;
  const int hi = lane >> 4;
  const int bcol = bhalf * 16 + lm;
  v8f acc = {};
  const int t0 = chunk * VT_PER_CHUNK;
  for (int t = 0; t < VT_PER_CHUNK; ++t) {
    const int i0 = (t0 + t) * 4;
    HFrag A, B;
    // A: row v=lm; K-chunk0 -> capsule i0+hi, d=0..7 ; chunk1 -> capsule i0+2+hi
    A.q[0] = *(const uint4*)(Wh + (((size_t)o * NCAPS + (i0 + hi))     * VDIM + lm) * DIM);
    A.q[1] = *(const uint4*)(Wh + (((size_t)o * NCAPS + (i0 + 2 + hi)) * VDIM + lm) * DIM);
    // B: col b=bcol; same K mapping; value = c[b,o,i] * u[b,i,d]
    const int ia = i0 + hi, ib = i0 + 2 + hi;
    const float ca = cmat[((size_t)bcol * OCAPS + o) * NCAPS + ia];
    const float cb = cmat[((size_t)bcol * OCAPS + o) * NCAPS + ib];
    const float* ua = u + ((size_t)bcol * NCAPS + ia) * DIM;
    const float* ub = u + ((size_t)bcol * NCAPS + ib) * DIM;
#pragma unroll
    for (int j = 0; j < 8; ++j) {
      B.h[j]     = (_Float16)(ca * ua[j]);
      B.h[j + 8] = (_Float16)(cb * ub[j]);
    }
    acc = wmma_f16(A, B, acc);
  }
  // lane owns column b=bcol, rows v = r + 8*hi
  float* sp = spart + ((size_t)chunk * BO + (size_t)bcol * OCAPS + o) * VDIM;
#pragma unroll
  for (int r = 0; r < 8; ++r) sp[r + 8 * hi] = acc[r];
}

// ---------------- reduce partials + squash -> v (or final output) ----------------
__global__ void k_squash_s(const float* __restrict__ spart, float* __restrict__ dst) {
  const int tid = blockIdx.x * blockDim.x + threadIdx.x;
  if (tid >= BO) return;
  float sv[VDIM];
#pragma unroll
  for (int v = 0; v < VDIM; ++v) sv[v] = 0.0f;
  for (int ch = 0; ch < VCHUNKS; ++ch) {
    const float* p = spart + ((size_t)ch * BO + tid) * VDIM;
#pragma unroll
    for (int v = 0; v < VDIM; ++v) sv[v] += p[v];
  }
  float sq = 0.0f;
#pragma unroll
  for (int v = 0; v < VDIM; ++v) sq += sv[v] * sv[v];
  const float scale = sq / ((1.0f + sq) * sqrtf(sq + 1e-8f));
#pragma unroll
  for (int v = 0; v < VDIM; ++v) dst[(size_t)tid * VDIM + v] = sv[v] * scale;
}

// ---------------- b-update: b[b,o,i] += sum_v v[b,o,v] * x_hat[b,o,i,v] (x_hat via WMMA)
// One zero-padded-K WMMA per capsule gives the x_hat[v,b] tile; dot with v via shfl pair.
__global__ void k_bupdate(const _Float16* __restrict__ Wh, const _Float16* __restrict__ uh,
                          const float* __restrict__ vvec, float* __restrict__ blog) {
  const int lane  = threadIdx.x & 31;
  const int bhalf = threadIdx.x >> 5;
  const int o = blockIdx.x;
  const int lm = lane & 15;
  const int hi = lane >> 4;
  const int b = bhalf * 16 + lm;
  float vv[8];
#pragma unroll
  for (int r = 0; r < 8; ++r) vv[r] = vvec[((size_t)b * OCAPS + o) * VDIM + r + 8 * hi];
  const uint4 z4 = make_uint4(0u, 0u, 0u, 0u);
  const int ibeg = blockIdx.y * I_PER_BCHUNK, iend = ibeg + I_PER_BCHUNK;
  for (int i = ibeg; i < iend; ++i) {
    HFrag A, B;
    const uint4 qa = *(const uint4*)(Wh + (((size_t)o * NCAPS + i) * VDIM + lm) * DIM);
    const uint4 qb = *(const uint4*)(uh + ((size_t)b * NCAPS + i) * DIM);
    A.q[0] = hi ? z4 : qa;   // K = 8..31 are zero padding
    A.q[1] = z4;
    B.q[0] = hi ? z4 : qb;
    B.q[1] = z4;
    v8f d = {};
    d = wmma_f16(A, B, d);
    float tp = 0.0f;
#pragma unroll
    for (int r = 0; r < 8; ++r) tp += vv[r] * d[r];   // rows r + 8*hi of column b
    tp += __shfl_xor(tp, 16, 32);                     // combine the two row-halves
    if (hi == 0) {
      const size_t idx = ((size_t)b * OCAPS + o) * NCAPS + i;
      blog[idx] += tp;                                // unique owner, no race
    }
  }
}

// ---------------- host side ----------------
extern "C" void kernel_launch(void* const* d_in, const int* in_sizes, int n_in,
                              void* d_out, int out_size, void* d_ws, size_t ws_size,
                              hipStream_t stream) {
  const float* x  = (const float*)d_in[0];
  const float* w1 = (const float*)d_in[1];
  const float* b1 = (const float*)d_in[2];
  const float* w2 = (const float*)d_in[3];
  const float* b2 = (const float*)d_in[4];
  const float* Wd = (const float*)d_in[5];
  float* out = (float*)d_out;
  char* ws = (char*)d_ws;

  // 256B-aligned workspace layout
  constexpr size_t SZ_W1H  = (size_t)OC1 * K1P * 2;                   //    196,608
  constexpr size_t SZ_W2H  = (size_t)OC2 * K2 * 2;                    // 10,616,832
  constexpr size_t SZ_WDH  = (size_t)OCAPS * NCAPS * VDIM * DIM * 2;  // 67,018,752
  constexpr size_t SZ_H16  = (size_t)BATCH * OC1 * OH1 * OH1 * 2;     // 18,939,904
  constexpr size_t SZ_P    = (size_t)BATCH * K2 * 4;                  //  2,654,208
  constexpr size_t SZ_U    = SZ_P;
  constexpr size_t SZ_UH   = SZ_P / 2;
  constexpr size_t SZ_BLOG = (size_t)BATCH * OCAPS * NCAPS * 4;       // 33,509,376
  constexpr size_t SZ_C    = SZ_BLOG;
  constexpr size_t SZ_SP   = (size_t)VCHUNKS * BO * VDIM * 4;         //  4,964,352
  constexpr size_t SZ_V    = (size_t)BO * VDIM * 4;
  constexpr size_t SZ_OF1  = (size_t)K1P * 4;                         //      1,536
  constexpr size_t SZ_OF2  = (size_t)K2 * 4;                          //     82,944

  size_t off = 0;
  _Float16* w1h  = (_Float16*)(ws + off); off += SZ_W1H;
  _Float16* w2h  = (_Float16*)(ws + off); off += SZ_W2H;
  _Float16* Wdh  = (_Float16*)(ws + off); off += SZ_WDH;
  _Float16* h16  = (_Float16*)(ws + off); off += SZ_H16;
  float*    p    = (float*)   (ws + off); off += SZ_P;
  float*    u    = (float*)   (ws + off); off += SZ_U;
  _Float16* uh   = (_Float16*)(ws + off); off += SZ_UH;
  float*    blog = (float*)   (ws + off); off += SZ_BLOG;
  float*    cmat = (float*)   (ws + off); off += SZ_C;
  float*    sp   = (float*)   (ws + off); off += SZ_SP;
  float*    vbuf = (float*)   (ws + off); off += SZ_V;
  int*      of1  = (int*)     (ws + off); off += SZ_OF1;
  int*      of2  = (int*)     (ws + off); off += SZ_OF2;
  (void)ws_size; (void)in_sizes; (void)n_in; (void)out_size;

  // setup: f16 weights + im2col offset tables
  k_cvt_w1<<<(OC1 * K1P + 255) / 256, 256, 0, stream>>>(w1, w1h);
  k_f32_to_f16<<<2048, 256, 0, stream>>>(w2, w2h, OC2 * K2);
  k_f32_to_f16<<<4096, 256, 0, stream>>>(Wd, Wdh, OCAPS * NCAPS * VDIM * DIM);
  k_make_offs<<<(K1P + K2 + 255) / 256, 256, 0, stream>>>(of1, of2);

  // conv1 + ReLU (f16 out), pconv, squash to capsules
  k_conv1<<<NT1, 128, 0, stream>>>(x, w1h, b1, of1, h16);
  k_pconv<<<NT2, 128, 0, stream>>>(h16, w2h, b2, of2, p);
  k_squash_u<<<(BI + 255) / 256, 256, 0, stream>>>(p, u, uh);

  // dynamic routing (3 iterations), x_hat recomputed via WMMA every pass
  k_zero_f32<<<2048, 256, 0, stream>>>(blog, BATCH * OCAPS * NCAPS);
  for (int it = 0; it < 3; ++it) {
    k_softmax_o<<<(BI + 255) / 256, 256, 0, stream>>>(blog, cmat);
    k_vpass<<<dim3(OCAPS, VCHUNKS), 64, 0, stream>>>(Wdh, u, cmat, sp);
    float* vdst = (it == 2) ? out : vbuf;
    k_squash_s<<<(BO + 255) / 256, 256, 0, stream>>>(sp, vdst);
    if (it < 2)
      k_bupdate<<<dim3(OCAPS, BCHUNKS), 64, 0, stream>>>(Wdh, uh, vbuf, blog);
  }
}
